// TransformerEncoderLayer_17128329577098
// MI455X (gfx1250) — compile-verified, matrix operand order checked by reading, not run on hardware
//
#include <hip/hip_runtime.h>
#include <hip/hip_bf16.h>

// ---------------------------------------------------------------------------
// Types / helpers for CDNA5 WMMA (wave32, 16x16x32 bf16 -> f32)
// ---------------------------------------------------------------------------
typedef __bf16 bf16_t;
typedef bf16_t v16bf __attribute__((ext_vector_type(16)));
typedef float  v8f   __attribute__((ext_vector_type(8)));
typedef unsigned int  v4u __attribute__((ext_vector_type(4)));
typedef int           v4i __attribute__((ext_vector_type(4)));
typedef int           v8i __attribute__((ext_vector_type(8)));

struct U32x8 { uint4 lo; uint4 hi; };
static_assert(sizeof(U32x8) == 32, "");
static_assert(sizeof(v16bf) == 32, "");

__device__ inline unsigned short f2bf(float f) {
    unsigned u = __builtin_bit_cast(unsigned, f);
    u += 0x7fffu + ((u >> 16) & 1u);           // round-to-nearest-even
    return (unsigned short)(u >> 16);
}
__device__ inline unsigned pk2(float a, float b) {
    return (unsigned)f2bf(a) | ((unsigned)f2bf(b) << 16);
}
// Fragment load from an LDS bf16 tile: row-major, elems 0..7 at koff, 8..15 at koff+16
// (matches ISA 16-bit A/B layout with koff = (lane>>4)*8 [+ k-chunk base]).
__device__ inline v16bf lds_frag(const unsigned short* smem, int row, int ldt, int koff) {
    const unsigned short* p = smem + row * ldt + koff;
    U32x8 u;
    u.lo = *(const uint4*)(p);
    u.hi = *(const uint4*)(p + 16);
    return __builtin_bit_cast(v16bf, u);
}
// Fragment load directly from a global bf16 row (contiguous K).
__device__ inline v16bf gbl_frag(const unsigned short* p, int koff) {
    U32x8 u;
    u.lo = *(const uint4*)(p + koff);
    u.hi = *(const uint4*)(p + koff + 16);
    return __builtin_bit_cast(v16bf, u);
}
__device__ inline v8f wmma_bf16(v16bf a, v16bf b, v8f c) {
    return __builtin_amdgcn_wmma_f32_16x16x32_bf16(false, a, false, b, (short)0, c, false, false);
}

// ---------------------------------------------------------------------------
// Tensor Data Mover: 2-D bf16 tile (global, row-major, stride_elems) -> LDS
// with per-row padding so LDS pitch = tile_cols + pad dwords.
// pad_int_code: interval = 2<<code dwords ; pad_amt_code: amount = code+1 dwords
// ---------------------------------------------------------------------------
#if __has_builtin(__builtin_amdgcn_tensor_load_to_lds)
#define USE_TDM 1
#endif

__device__ inline void tdm_wait0() {
#if __has_builtin(__builtin_amdgcn_s_wait_tensorcnt)
    __builtin_amdgcn_s_wait_tensorcnt((unsigned short)0);
#else
    asm volatile("s_wait_tensorcnt 0x0" ::: "memory");
#endif
}

#ifdef USE_TDM
__device__ inline void tdm_load_2d(unsigned lds_off, const void* gp,
                                   unsigned tile_cols, unsigned tile_rows,
                                   unsigned stride_elems,
                                   unsigned pad_int_code, unsigned pad_amt_code,
                                   unsigned dim0, unsigned dim1) {
    unsigned long long ga = (unsigned long long)(__UINTPTR_TYPE__)gp;
    v4u g0;
    g0.x = 1u;                                           // count=1 (valid user D#)
    g0.y = lds_off;                                      // lds_addr (bytes)
    g0.z = (unsigned)ga;                                 // global_addr[31:0]
    g0.w = (unsigned)((ga >> 32) & 0x1FFFFFFull) | (2u << 30);  // addr[56:32], type=2
    unsigned w0 = (1u << 16)                             // data_size = 2 bytes
                | (1u << 20)                             // pad_enable
                | (pad_int_code << 22)
                | (pad_amt_code << 25);
    unsigned w1 = (dim0 & 0xFFFFu) << 16;                // tensor_dim0[15:0]
    unsigned w2 = ((dim0 >> 16) & 0xFFFFu) | ((dim1 & 0xFFFFu) << 16);
    unsigned w3 = ((dim1 >> 16) & 0xFFFFu) | (tile_cols << 16);   // tile_dim0
    unsigned w4 = tile_rows & 0xFFFFu;                   // tile_dim1 (tile_dim2 = 0)
    unsigned w5 = stride_elems;                          // tensor_dim0_stride[31:0]
    unsigned w6 = 0u, w7 = 0u;
    v8i g1 = { (int)w0, (int)w1, (int)w2, (int)w3, (int)w4, (int)w5, (int)w6, (int)w7 };
    v4i z4 = { 0, 0, 0, 0 };
#if defined(__clang_major__) && (__clang_major__ >= 23)
    v8i z8 = { 0, 0, 0, 0, 0, 0, 0, 0 };
    __builtin_amdgcn_tensor_load_to_lds(g0, g1, z4, z4, z8, 0);
#else
    __builtin_amdgcn_tensor_load_to_lds(g0, g1, z4, z4, 0);
#endif
}
#endif

__device__ inline unsigned lds_off32(const void* p) {
    return (unsigned)(__UINTPTR_TYPE__)p;                // low 32 bits = LDS byte offset
}

// ---------------------------------------------------------------------------
// f32 -> bf16 conversion pre-pass (8 elements / thread)
// ---------------------------------------------------------------------------
__global__ __launch_bounds__(256) void cvt_kernel(const float* __restrict__ x,
                                                  unsigned short* __restrict__ y, int n) {
    int i = (blockIdx.x * 256 + threadIdx.x) * 8;
    if (i >= n) return;
    float4 a = *(const float4*)(x + i);
    float4 b = *(const float4*)(x + i + 4);
    uint4 o;
    o.x = pk2(a.x, a.y); o.y = pk2(a.z, a.w);
    o.z = pk2(b.x, b.y); o.w = pk2(b.z, b.w);
    *(uint4*)(y + i) = o;
}

// ---------------------------------------------------------------------------
// GEMM: C[M,N] = act(A[M,K] * W[N,K]^T + bias [+ res]); A,W bf16; C f32 or bf16.
// Block tile 128x128x32, 8 waves x (32x64). Double-buffered tiles: the TDM
// loads tile i+1 while the waves run WMMA on tile i; one barrier / K-step.
// ---------------------------------------------------------------------------
template <bool RELU, bool RES, bool OUTBF>
__global__ __launch_bounds__(256) void gemm_kernel(
    const unsigned short* __restrict__ A, const unsigned short* __restrict__ W,
    const float* __restrict__ bias, const float* __restrict__ res,
    void* __restrict__ Cout, int M, int N, int K, float oscale)
{
    __shared__ unsigned short As[2][128 * 40];   // 32 bf16 data + 8 pad per row
    __shared__ unsigned short Bs[2][128 * 40];

    const int tid = threadIdx.x, lane = tid & 31;
    const int wave = tid >> 5;
    const int wm = (wave & 3) * 32, wn = (wave >> 2) * 64;
    const int bm = blockIdx.y * 128, bn = blockIdx.x * 128;
    const int rowin = lane & 15, hf = lane >> 4, kb = hf * 8;
    const bool issuer = (__builtin_amdgcn_readfirstlane(tid) < 32);   // wave 0
    const int nk = K >> 5;

#ifndef USE_TDM
    uint4 pa[2], pb[2];
    const int srow = tid >> 2, sc8 = (tid & 3) * 8;      // thread's 2 chunks: rows srow, srow+64
#endif

    // issue staging for K-step `it` into buffer `buf`
    auto stage_issue = [&](int it, int buf) {
#ifdef USE_TDM
        if (issuer) {
            tdm_load_2d(lds_off32(As[buf]), A + (size_t)bm * K + it * 32,
                        32u, 128u, (unsigned)K, 3u /*16 dw*/, 3u /*4 dw*/,
                        (unsigned)K, (unsigned)M);
            tdm_load_2d(lds_off32(Bs[buf]), W + (size_t)bn * K + it * 32,
                        32u, 128u, (unsigned)K, 3u, 3u,
                        (unsigned)K, (unsigned)N);
        }
#else
#pragma unroll
        for (int s = 0; s < 2; ++s) {
            int row = srow + s * 64;
            pa[s] = *(const uint4*)(A + (size_t)(bm + row) * K + it * 32 + sc8);
            pb[s] = *(const uint4*)(W + (size_t)(bn + row) * K + it * 32 + sc8);
        }
#endif
    };
    // make staged data resident in LDS buffer `buf`
    auto stage_finish = [&](int buf) {
#ifdef USE_TDM
        if (issuer) tdm_wait0();
#else
#pragma unroll
        for (int s = 0; s < 2; ++s) {
            int row = srow + s * 64;
            *(uint4*)(As[buf] + row * 40 + sc8) = pa[s];
            *(uint4*)(Bs[buf] + row * 40 + sc8) = pb[s];
        }
#endif
    };

    const v8f vzero = {0.f, 0.f, 0.f, 0.f, 0.f, 0.f, 0.f, 0.f};
    v8f acc[2][4];
#pragma unroll
    for (int i = 0; i < 2; ++i)
#pragma unroll
        for (int j = 0; j < 4; ++j) acc[i][j] = vzero;

    stage_issue(0, 0);
    stage_finish(0);
    __syncthreads();

    for (int it = 0; it < nk; ++it) {
        const int cur = it & 1, nxt = cur ^ 1;
        if (it + 1 < nk) stage_issue(it + 1, nxt);     // DMA runs under the WMMAs

        v16bf af[2], bfr[4];
#pragma unroll
        for (int i = 0; i < 2; ++i) af[i] = lds_frag(As[cur], wm + i * 16 + rowin, 40, kb);
#pragma unroll
        for (int j = 0; j < 4; ++j) bfr[j] = lds_frag(Bs[cur], wn + j * 16 + rowin, 40, kb);
#pragma unroll
        for (int i = 0; i < 2; ++i)
#pragma unroll
            for (int j = 0; j < 4; ++j)
                acc[i][j] = wmma_bf16(af[i], bfr[j], acc[i][j]);

        if (it + 1 < nk) stage_finish(nxt);
        __syncthreads();
    }

    // Epilogue: C layout -> lane holds col = lane&15, VGPR r -> row r + 8*(lane>>4).
#pragma unroll
    for (int j = 0; j < 4; ++j) {
        int gcol = bn + wn + j * 16 + rowin;
        float bc = bias[gcol];
#pragma unroll
        for (int i = 0; i < 2; ++i) {
#pragma unroll
            for (int r = 0; r < 8; ++r) {
                int grow = bm + wm + i * 16 + r + 8 * hf;
                size_t idx = (size_t)grow * N + gcol;
                float vv = acc[i][j][r] + bc;
                if (RES)  vv += res[idx];
                if (RELU) vv = fmaxf(vv, 0.f);
                if (OUTBF) ((unsigned short*)Cout)[idx] = f2bf(vv * oscale);
                else       ((float*)Cout)[idx] = vv;
            }
        }
    }
}

// ---------------------------------------------------------------------------
// Flash attention: grid (S/128, NH, B), 8 waves; wave = one 16-query tile.
// q,k,v: bf16 [B*S, NH*DH] (q pre-scaled by 1/8). ctx out: bf16.
// Double-buffered K (TDM) and V (VGPR-staged transpose); one barrier / chunk.
// ---------------------------------------------------------------------------
__global__ __launch_bounds__(256) void attn_kernel(
    const unsigned short* __restrict__ q, const unsigned short* __restrict__ k,
    const unsigned short* __restrict__ v, const float* __restrict__ npm,
    unsigned short* __restrict__ ctx)
{
    __shared__ unsigned short Kt[2][32 * 72];   // [key][dh] : 64 data + 8 pad
    __shared__ unsigned short Vt[2][64 * 40];   // [dh][key] : transposed
    __shared__ unsigned short Pt[8 * 16 * 40];  // per-wave probs [qrow][key]

    const int tid = threadIdx.x, lane = tid & 31, wave = tid >> 5;
    const int b = blockIdx.z, h = blockIdx.y;
    const int q0 = blockIdx.x * 128 + wave * 16;
    const int rowin = lane & 15, hf = lane >> 4, kb = hf * 8;
    const bool issuer = (__builtin_amdgcn_readfirstlane(tid) < 32);   // wave 0
    const int skey = tid >> 3, sc8 = (tid & 7) * 8;     // staging coords

    uint4 vreg;
#ifndef USE_TDM
    uint4 kreg;
#endif
    auto stage_issue = [&](int kc, int buf) {
        size_t base = (size_t)(b * 1024 + kc + skey) * 768 + h * 64 + sc8;
#ifdef USE_TDM
        (void)buf;
        if (issuer)
            tdm_load_2d(lds_off32(Kt[buf]), k + (size_t)(b * 1024 + kc) * 768 + h * 64,
                        64u, 32u, 768u, 4u /*32 dw*/, 3u /*4 dw*/, 768u, 1u << 20);
#else
        kreg = *(const uint4*)(k + base);
#endif
        vreg = *(const uint4*)(v + base);
    };
    auto stage_finish = [&](int buf) {
#ifndef USE_TDM
        *(uint4*)(Kt[buf] + skey * 72 + sc8) = kreg;
#endif
        unsigned vals[4] = {vreg.x, vreg.y, vreg.z, vreg.w};
#pragma unroll
        for (int e = 0; e < 4; ++e) {          // transpose into Vt[dh][key]
            Vt[buf][(sc8 + 2 * e + 0) * 40 + skey] = (unsigned short)(vals[e] & 0xFFFFu);
            Vt[buf][(sc8 + 2 * e + 1) * 40 + skey] = (unsigned short)(vals[e] >> 16);
        }
#ifdef USE_TDM
        if (issuer) tdm_wait0();
#endif
    };

    // Q fragments load straight from global bf16 (dh 0..31, 32..63).
    v16bf qa[2];
    {
        const unsigned short* qrow = q + (size_t)(b * 1024 + q0 + rowin) * 768 + h * 64;
        qa[0] = gbl_frag(qrow, kb);
        qa[1] = gbl_frag(qrow + 32, kb);
    }

    const v8f vzero = {0.f, 0.f, 0.f, 0.f, 0.f, 0.f, 0.f, 0.f};
    float m[8], l[8];
    v8f cacc[4];
#pragma unroll
    for (int r = 0; r < 8; ++r) { m[r] = -3.0e38f; l[r] = 0.f; }
#pragma unroll
    for (int g = 0; g < 4; ++g) cacc[g] = vzero;

    unsigned short* Pw = Pt + wave * 16 * 40;

    stage_issue(0, 0);
    stage_finish(0);
    __syncthreads();

    for (int it = 0; it < 32; ++it) {
        const int kc = it * 32;
        const int cur = it & 1, nxt = cur ^ 1;
        if (it + 1 < 32) stage_issue(kc + 32, nxt);   // prefetch next chunk

        // scores: two 16-key tiles, K-dim 64 = 2 wmma each.
        v8f sc[2];
#pragma unroll
        for (int t = 0; t < 2; ++t) {
            sc[t] = vzero;
#pragma unroll
            for (int c = 0; c < 2; ++c) {
                v16bf kf = lds_frag(Kt[cur], t * 16 + rowin, 72, c * 32 + kb);
                sc[t] = wmma_bf16(qa[c], kf, sc[t]);
            }
            float mval = npm[b * 1024 + kc + t * 16 + rowin];    // key pad mask
#pragma unroll
            for (int r = 0; r < 8; ++r)
                sc[t][r] = (mval != 0.f) ? sc[t][r] : -3.0e38f;
        }

        // Online softmax per row (rows live across 16-lane halves).
#pragma unroll
        for (int r = 0; r < 8; ++r) {
            float cm = fmaxf(sc[0][r], sc[1][r]);
#pragma unroll
            for (int off = 8; off > 0; off >>= 1)
                cm = fmaxf(cm, __shfl_xor(cm, off, 16));
            float mn = fmaxf(m[r], cm);
            float scale = __expf(m[r] - mn);
            float p0 = __expf(sc[0][r] - mn);
            float p1 = __expf(sc[1][r] - mn);
            float srow = p0 + p1;
#pragma unroll
            for (int off = 8; off > 0; off >>= 1)
                srow += __shfl_xor(srow, off, 16);
            l[r] = l[r] * scale + srow;
            m[r] = mn;
#pragma unroll
            for (int g = 0; g < 4; ++g) cacc[g][r] *= scale;
            int prow = r + 8 * hf;
            Pw[prow * 40 + rowin]      = f2bf(p0);
            Pw[prow * 40 + 16 + rowin] = f2bf(p1);
        }

        // ctx += P(16x32) * V(32x64). Same-wave DS ops are in-order: no barrier.
        v16bf pf = lds_frag(Pw, rowin, 40, kb);
#pragma unroll
        for (int g = 0; g < 4; ++g) {
            v16bf vf = lds_frag(Vt[cur], g * 16 + rowin, 40, kb);
            cacc[g] = wmma_bf16(pf, vf, cacc[g]);
        }

        if (it + 1 < 32) stage_finish(nxt);
        __syncthreads();
    }

    // Normalize and write ctx as bf16 (feeds the out-projection GEMM).
#pragma unroll
    for (int r = 0; r < 8; ++r) {
        float inv = 1.0f / l[r];
        int qrow = q0 + r + 8 * hf;
        size_t base = (size_t)(b * 1024 + qrow) * 768 + h * 64;
#pragma unroll
        for (int g = 0; g < 4; ++g)
            ctx[base + g * 16 + rowin] = f2bf(cacc[g][r] * inv);
    }
}

// ---------------------------------------------------------------------------
// LayerNorm over H=768 + pad mask; optionally also emit a bf16 copy.
// ---------------------------------------------------------------------------
template <bool WB>
__global__ __launch_bounds__(256) void ln_kernel(
    const float* __restrict__ x, const float* __restrict__ g,
    const float* __restrict__ bta, const float* __restrict__ npm,
    float* __restrict__ y, unsigned short* __restrict__ yb)
{
    __shared__ float rs[256], rq[256];
    __shared__ float s_mu, s_rstd;
    const int row = blockIdx.x, tid = threadIdx.x;
    const float* xr = x + (size_t)row * 768;
    float s = 0.f, sq = 0.f;
    for (int c = tid; c < 768; c += 256) { float vv = xr[c]; s += vv; sq += vv * vv; }
    rs[tid] = s; rq[tid] = sq;
    __syncthreads();
    for (int o = 128; o > 0; o >>= 1) {
        if (tid < o) { rs[tid] += rs[tid + o]; rq[tid] += rq[tid + o]; }
        __syncthreads();
    }
    if (tid == 0) {
        float mu  = rs[0] * (1.f / 768.f);
        float var = rq[0] * (1.f / 768.f) - mu * mu;
        s_mu = mu; s_rstd = rsqrtf(var + 1e-5f);
    }
    __syncthreads();
    const float mu = s_mu, rstd = s_rstd, mask = npm[row];
    for (int c = tid; c < 768; c += 256) {
        float o = ((xr[c] - mu) * rstd * g[c] + bta[c]) * mask;
        y[(size_t)row * 768 + c] = o;
        if (WB) yb[(size_t)row * 768 + c] = f2bf(o);
    }
}

// ---------------------------------------------------------------------------
// Orchestration
// ---------------------------------------------------------------------------
extern "C" void kernel_launch(void* const* d_in, const int* in_sizes, int n_in,
                              void* d_out, int out_size, void* d_ws, size_t ws_size,
                              hipStream_t stream)
{
    (void)in_sizes; (void)n_in; (void)out_size; (void)ws_size;
    const float* token = (const float*)d_in[0];
    // d_in[1] = pad_mask (unused; key validity derived from no_pad_mask)
    const float* npm = (const float*)d_in[2];
    const float* wq = (const float*)d_in[3];  const float* bq = (const float*)d_in[4];
    const float* wk = (const float*)d_in[5];  const float* bk = (const float*)d_in[6];
    const float* wv = (const float*)d_in[7];  const float* bv = (const float*)d_in[8];
    const float* wo = (const float*)d_in[9];  const float* bo = (const float*)d_in[10];
    const float* g1 = (const float*)d_in[11]; const float* b1n = (const float*)d_in[12];
    const float* w1 = (const float*)d_in[13]; const float* b1 = (const float*)d_in[14];
    const float* w2 = (const float*)d_in[15]; const float* b2 = (const float*)d_in[16];
    const float* g2 = (const float*)d_in[17]; const float* b2n = (const float*)d_in[18];

    typedef unsigned short u16;
    const size_t MH  = (size_t)8192 * 768;     // 6.29 M elements
    const size_t WSQ = (size_t)768 * 768;      // square weight
    const size_t WFF = (size_t)768 * 3072;     // FFN weight

    char* wp = (char*)d_ws;
    float* tmp   = (float*)wp;  wp += MH * 4;  // pre-LN buffers (f32, residual-added)
    float* attnf = (float*)wp;  wp += MH * 4;  // LN1 output, f32 (FFN2 residual)
    u16* qb     = (u16*)wp;     wp += MH * 2;
    u16* kbb    = (u16*)wp;     wp += MH * 2;
    u16* vbb    = (u16*)wp;     wp += MH * 2;
    u16* ctxb   = (u16*)wp;     wp += MH * 2;
    u16* tokenb = (u16*)wp;     wp += MH * 2;
    u16* attnb  = (u16*)wp;     wp += MH * 2;  // LN1 output, bf16 (FFN1 input)
    u16* wqb    = (u16*)wp;     wp += WSQ * 2;
    u16* wkbuf  = (u16*)wp;     wp += WSQ * 2;
    u16* wvbuf  = (u16*)wp;     wp += WSQ * 2;
    u16* wob    = (u16*)wp;     wp += WSQ * 2;
    u16* w1b    = (u16*)wp;     wp += WFF * 2;
    u16* w2b    = (u16*)wp;     wp += WFF * 2;
    u16* h1b    = qb;   // FFN hidden (8192x3072) aliases q,k,v,ctx: all dead by then
    float* out  = (float*)d_out;

    dim3 blk(256);
    // bf16 pre-conversion of activations-once and all weights
    cvt_kernel<<<dim3((unsigned)(MH / 2048)),  blk, 0, stream>>>(token, tokenb, (int)MH);
    cvt_kernel<<<dim3((unsigned)(WSQ / 2048)), blk, 0, stream>>>(wq, wqb,   (int)WSQ);
    cvt_kernel<<<dim3((unsigned)(WSQ / 2048)), blk, 0, stream>>>(wk, wkbuf, (int)WSQ);
    cvt_kernel<<<dim3((unsigned)(WSQ / 2048)), blk, 0, stream>>>(wv, wvbuf, (int)WSQ);
    cvt_kernel<<<dim3((unsigned)(WSQ / 2048)), blk, 0, stream>>>(wo, wob,   (int)WSQ);
    cvt_kernel<<<dim3((unsigned)(WFF / 2048)), blk, 0, stream>>>(w1, w1b,   (int)WFF);
    cvt_kernel<<<dim3((unsigned)(WFF / 2048)), blk, 0, stream>>>(w2, w2b,   (int)WFF);

    // QKV projections (bf16 out; Q pre-scaled by 1/sqrt(DH) = 1/8)
    gemm_kernel<false, false, true><<<dim3(6, 64), blk, 0, stream>>>(tokenb, wqb,   bq, nullptr, qb,  8192, 768, 768, 0.125f);
    gemm_kernel<false, false, true><<<dim3(6, 64), blk, 0, stream>>>(tokenb, wkbuf, bk, nullptr, kbb, 8192, 768, 768, 1.0f);
    gemm_kernel<false, false, true><<<dim3(6, 64), blk, 0, stream>>>(tokenb, wvbuf, bv, nullptr, vbb, 8192, 768, 768, 1.0f);
    // Attention
    attn_kernel<<<dim3(8, 12, 8), blk, 0, stream>>>(qb, kbb, vbb, npm, ctxb);
    // Out projection + residual (f32), LN1 + mask (f32 + bf16 copies)
    gemm_kernel<false, true, false><<<dim3(6, 64), blk, 0, stream>>>(ctxb, wob, bo, token, tmp, 8192, 768, 768, 1.0f);
    ln_kernel<true><<<dim3(8192), blk, 0, stream>>>(tmp, g1, b1n, npm, attnf, attnb);
    // FFN
    gemm_kernel<true,  false, true ><<<dim3(24, 64), blk, 0, stream>>>(attnb, w1b, b1, nullptr, h1b, 8192, 3072, 768, 1.0f);
    gemm_kernel<false, true,  false><<<dim3(6, 64),  blk, 0, stream>>>(h1b, w2b, b2, attnf, tmp, 8192, 768, 3072, 1.0f);
    ln_kernel<false><<<dim3(8192), blk, 0, stream>>>(tmp, g2, b2n, npm, out, nullptr);
}
